// Agent4_47296179863718
// MI455X (gfx1250) — compile-verified, matrix operand order checked by reading, not run on hardware
//
#include <hip/hip_runtime.h>
#include <hip/hip_bf16.h>
#include <stdint.h>

#define BN   32768
#define DN   8
#define LN   16
#define TAB  524288
#define HID  512
#define KPAD 64

typedef __attribute__((ext_vector_type(16))) __bf16 bf16x16;
typedef __attribute__((ext_vector_type(8)))  float  f32x8;

union FragAB { bf16x16 v; uint4 q[2]; };

__device__ __forceinline__ unsigned short f2bf(float f) {
  unsigned int u = __float_as_uint(f);
  unsigned int r = u + 0x7FFFu + ((u >> 16) & 1u);   // round-to-nearest-even
  return (unsigned short)(r >> 16);
}

// ---------------------------------------------------------------------------
// Weight prep: fp32 -> bf16, pad layer1 K 40->64, merge act/val heads into W3
// ---------------------------------------------------------------------------
__global__ __launch_bounds__(256) void prep_weights(
    const float* __restrict__ l1w, const float* __restrict__ l2w,
    const float* __restrict__ actw, const float* __restrict__ actb,
    const float* __restrict__ valw, const float* __restrict__ valb,
    unsigned short* __restrict__ W1, unsigned short* __restrict__ W2,
    unsigned short* __restrict__ W3, float* __restrict__ B3)
{
  const int gid = blockIdx.x * 256 + threadIdx.x;    // grid covers 512*512
  W2[gid] = f2bf(l2w[gid]);
  if (gid < HID * KPAD) {
    const int n = gid >> 6, k = gid & 63;
    W1[gid] = (k < 40) ? f2bf(l1w[n * 40 + k]) : (unsigned short)0;
  }
  if (gid < 16 * HID) {
    const int n = gid >> 9, k = gid & 511;
    const float v = (n < 4) ? actw[n * HID + k] : (n == 4 ? valw[k] : 0.0f);
    W3[gid] = f2bf(v);
  }
  if (gid < 16) B3[gid] = (gid < 4) ? actb[gid] : (gid == 4 ? valb[0] : 0.0f);
}

// ---------------------------------------------------------------------------
// Hash encoding: one thread per (b, l) pair
// ---------------------------------------------------------------------------
__global__ __launch_bounds__(256) void encode_kernel(
    const float* __restrict__ x, const float* __restrict__ T,
    const float* __restrict__ Kv, unsigned short* __restrict__ hA)
{
  const unsigned int PR[8] = {2654436881u, 1620619981u, 1500450271u, 3267000013u,
                              1459886047u, 4093082899u,  986956175u, 3628273133u};
  const int gid = blockIdx.x * 256 + threadIdx.x;
  const int l = gid & (LN - 1);
  const int b = gid >> 4;
  const float kl = Kv[l];
  const float* xr = x + (size_t)b * DN;

  float flo[8], fr[8];
  float ps = 0.0f;
#pragma unroll
  for (int d = 0; d < 8; ++d) {
    ps += xr[d];
    const float pd = ps * kl;
    flo[d] = floorf(pd);
    fr[d]  = pd - truncf(pd);
  }
  float c[8];
#pragma unroll
  for (int d = 0; d < 8; ++d) c[d] = fr[d];
  // Batcher odd-even merge sort, 19 comparators, constant indices (registers)
#define CAS(i,j) { const float lo_ = fminf(c[i], c[j]); const float hi_ = fmaxf(c[i], c[j]); c[i] = lo_; c[j] = hi_; }
  CAS(0,1) CAS(2,3) CAS(4,5) CAS(6,7)
  CAS(0,2) CAS(1,3) CAS(4,6) CAS(5,7)
  CAS(1,2) CAS(5,6)
  CAS(0,4) CAS(1,5) CAS(2,6) CAS(3,7)
  CAS(2,4) CAS(3,5)
  CAS(1,2) CAS(3,4) CAS(5,6)
#undef CAS

  const float* Tl = T + (size_t)l * (TAB * 2);
  float f0 = 0.0f, f1 = 0.0f, cprev = 0.0f;
#pragma unroll
  for (int j = 0; j < 9; ++j) {
    const float cj = (j < 8) ? c[j] : 1.0f;
    unsigned int h = 0;
    int aprev = 0;
#pragma unroll
    for (int d = 0; d < 8; ++d) {
      const int acum = (int)flo[d] + (((j < 8) && (fr[d] >= cj)) ? 1 : 0);
      h ^= (unsigned int)(acum - aprev) * PR[d];
      aprev = acum;
    }
    const float* t = Tl + (size_t)(h & (TAB - 1)) * 2;   // TABLE is 2^19: & == floored mod
    const float w = cj - cprev;
    cprev = cj;
    f0 += t[0] * w;
    f1 += t[1] * w;
  }
  f0 /= kl;
  f1 /= kl;

  unsigned short* hrow = hA + (size_t)b * KPAD;
  hrow[DN + l * 2]     = f2bf(f0);
  hrow[DN + l * 2 + 1] = f2bf(f1);
  if (l == 0) {
#pragma unroll
    for (int d = 0; d < 8; ++d) hrow[d] = f2bf(xr[d]);
  } else if (l == 1) {
#pragma unroll
    for (int i = 0; i < 12; ++i) hrow[40 + i] = 0;       // zero K padding
  } else if (l == 2) {
#pragma unroll
    for (int i = 0; i < 12; ++i) hrow[52 + i] = 0;
  }
}

// ---------------------------------------------------------------------------
// WMMA GEMM: Out(M x 512) = relu(A(M x K) * W(512 x K)^T + bias), bf16 in/out
// Block: 256 threads = 8 waves; BM = 128 rows.
// A K-chunks are double-buffered in LDS via CDNA5 async load-to-LDS
// (global_load_async_to_lds_b128, ASYNCcnt), overlapping staging with WMMA.
// ---------------------------------------------------------------------------
template <int KDIM, bool RELU>
__global__ __launch_bounds__(256) void gemm_bf16(
    const unsigned short* __restrict__ A, const unsigned short* __restrict__ W,
    const float* __restrict__ bias, unsigned short* __restrict__ Out)
{
  constexpr int N = HID, BM = 128, KC = 32, SAS = 40;   // SAS: padded LDS row pitch
  constexpr int NSTEP = KDIM / KC;
  __shared__ unsigned short sA[2][BM * SAS];            // 2 x 10 KB
  const int tid = threadIdx.x, wave = tid >> 5, lane = tid & 31;
  const int lr = lane & 15;     // row-in-subtile (A) / output column n (B, D)
  const int kh = lane >> 4;     // K-half selector per ISA 16-bit A/B layout
  const int m0 = blockIdx.x * BM;

  // Per-thread staging coordinates: 512 16-byte chunks over 256 threads
  const int rowA = tid >> 2,          kpA = (tid & 3) * 8;
  const int rowB = (tid + 256) >> 2,  kpB = ((tid + 256) & 3) * 8;

  auto stage = [&](int k0, int buf) {
    const unsigned short* g0 = A + (size_t)(m0 + rowA) * KDIM + k0 + kpA;
    const unsigned short* g1 = A + (size_t)(m0 + rowB) * KDIM + k0 + kpB;
    const unsigned int l0 = (unsigned int)(uintptr_t)&sA[buf][rowA * SAS + kpA];
    const unsigned int l1 = (unsigned int)(uintptr_t)&sA[buf][rowB * SAS + kpB];
    asm volatile("global_load_async_to_lds_b128 %0, %1, off"
                 :: "v"(l0), "v"(g0) : "memory");
    asm volatile("global_load_async_to_lds_b128 %0, %1, off"
                 :: "v"(l1), "v"(g1) : "memory");
  };

  for (int ngrp = 0; ngrp < N / 128; ++ngrp) {
    const int nb = ngrp * 128 + wave * 16;
    f32x8 acc[8];
#pragma unroll
    for (int s = 0; s < 8; ++s)
#pragma unroll
      for (int i = 0; i < 8; ++i) acc[s][i] = 0.0f;

    const unsigned short* wrow = W + (size_t)(nb + lr) * KDIM;

    __syncthreads();            // prior readers of buf0 are done
    stage(0, 0);

    for (int i = 0; i < NSTEP; ++i) {
      asm volatile("s_wait_asynccnt 0" ::: "memory");  // my buf(i) writes landed
      __syncthreads();          // everyone's landed; buf(i^1) readers finished
      if (i + 1 < NSTEP) stage((i + 1) * KC, (i + 1) & 1);

      const unsigned short* sbuf = sA[i & 1];
      const int k0 = i * KC;

      FragAB bf;  // weights as B: lane n = lr, per-lane K chunks mirror A layout
      bf.q[0] = *(const uint4*)(wrow + k0 + kh * 8);
      bf.q[1] = *(const uint4*)(wrow + k0 + kh * 8 + 16);
#pragma unroll
      for (int s = 0; s < 8; ++s) {
        FragAB af;
        const unsigned short* ar = sbuf + (s * 16 + lr) * SAS + kh * 8;
        af.q[0] = *(const uint4*)(ar);
        af.q[1] = *(const uint4*)(ar + 16);
        acc[s] = __builtin_amdgcn_wmma_f32_16x16x32_bf16(
            false, af.v, false, bf.v, (short)0, acc[s], false, false);
      }
    }
    // Epilogue: D layout -> lane holds n = lr, rows kh*8 + r within subtile
    const float bn = bias[nb + lr];
#pragma unroll
    for (int s = 0; s < 8; ++s) {
      const int rbase = m0 + s * 16 + kh * 8;
#pragma unroll
      for (int r = 0; r < 8; ++r) {
        float v = acc[s][r] + bn;
        if (RELU) v = fmaxf(v, 0.0f);
        Out[(size_t)(rbase + r) * N + nb + lr] = f2bf(v);
      }
    }
  }
}

// ---------------------------------------------------------------------------
// Heads: (action 4 + value 1) padded to N=16, K=512, fp32 outputs to d_out
// ---------------------------------------------------------------------------
__global__ __launch_bounds__(256) void heads_kernel(
    const unsigned short* __restrict__ H, const unsigned short* __restrict__ W3,
    const float* __restrict__ B3, float* __restrict__ out)
{
  const int tid = threadIdx.x, wave = tid >> 5, lane = tid & 31;
  const int lr = lane & 15, kh = lane >> 4;
  const int row0 = blockIdx.x * 128 + wave * 16;
  const unsigned short* ar = H + (size_t)(row0 + lr) * HID;
  const unsigned short* wr = W3 + (size_t)lr * HID;

  f32x8 acc;
#pragma unroll
  for (int i = 0; i < 8; ++i) acc[i] = 0.0f;

#pragma unroll 4
  for (int k0 = 0; k0 < HID; k0 += 32) {
    FragAB a, b;
    a.q[0] = *(const uint4*)(ar + k0 + kh * 8);
    a.q[1] = *(const uint4*)(ar + k0 + kh * 8 + 16);
    b.q[0] = *(const uint4*)(wr + k0 + kh * 8);
    b.q[1] = *(const uint4*)(wr + k0 + kh * 8 + 16);
    acc = __builtin_amdgcn_wmma_f32_16x16x32_bf16(
        false, a.v, false, b.v, (short)0, acc, false, false);
  }
  const float bb = B3[lr];
#pragma unroll
  for (int r = 0; r < 8; ++r) {
    const int row = row0 + kh * 8 + r;
    const float v = acc[r] + bb;
    if (lr < 4)       out[(size_t)row * 4 + lr] = v;        // action (B,4)
    else if (lr == 4) out[(size_t)BN * 4 + row] = v;        // value  (B,1)
  }
}

// ---------------------------------------------------------------------------
extern "C" void kernel_launch(void* const* d_in, const int* in_sizes, int n_in,
                              void* d_out, int out_size, void* d_ws, size_t ws_size,
                              hipStream_t stream) {
  const float* x    = (const float*)d_in[0];
  const float* T    = (const float*)d_in[1];
  const float* Kv   = (const float*)d_in[2];
  const float* l1w  = (const float*)d_in[3];
  const float* l1b  = (const float*)d_in[4];
  const float* l2w  = (const float*)d_in[5];
  const float* l2b  = (const float*)d_in[6];
  const float* actw = (const float*)d_in[7];
  const float* actb = (const float*)d_in[8];
  const float* valw = (const float*)d_in[9];
  const float* valb = (const float*)d_in[10];
  (void)in_sizes; (void)n_in; (void)out_size; (void)ws_size;

  char* wsp = (char*)d_ws;
  size_t o = 0;
  auto carve = [&](size_t bytes) {
    void* p = wsp + o;
    o += (bytes + 255) & ~(size_t)255;
    return p;
  };
  unsigned short* hA = (unsigned short*)carve((size_t)BN * KPAD * 2);   // 4 MB
  unsigned short* W1 = (unsigned short*)carve((size_t)HID * KPAD * 2);  // 64 KB
  unsigned short* W2 = (unsigned short*)carve((size_t)HID * HID * 2);   // 512 KB
  unsigned short* W3 = (unsigned short*)carve((size_t)16 * HID * 2);    // 16 KB
  float*          B3 = (float*)carve(16 * 4);
  unsigned short* H1 = (unsigned short*)carve((size_t)BN * HID * 2);    // 32 MB
  unsigned short* H2 = (unsigned short*)carve((size_t)BN * HID * 2);    // 32 MB

  prep_weights<<<(HID * HID) / 256, 256, 0, stream>>>(
      l1w, l2w, actw, actb, valw, valb, W1, W2, W3, B3);
  encode_kernel<<<(BN * LN) / 256, 256, 0, stream>>>(x, T, Kv, hA);
  gemm_bf16<KPAD, true><<<BN / 128, 256, 0, stream>>>(hA, W1, l1b, H1);
  gemm_bf16<HID,  true><<<BN / 128, 256, 0, stream>>>(H1, W2, l2b, H2);
  heads_kernel<<<BN / 128, 256, 0, stream>>>(H2, W3, B3, (float*)d_out);
}
